// RTLSTM_Hawkes_3152505995669
// MI455X (gfx1250) — matrix operand order repaired and checked by reading, not compiled, not run on hardware
//
#include <hip/hip_runtime.h>

#define B_  256
#define S_  512
#define D_  256
#define H_  256
#define H3  768
#define H4  1024
#define BT  16
#define TPB 512   // 16 waves: 4 waves per SIMD32 in the WGP for latency hiding
#define ATPB 256

typedef __attribute__((ext_vector_type(16))) _Float16 v16h;
typedef __attribute__((ext_vector_type(8)))  _Float16 v8h;
typedef __attribute__((ext_vector_type(8)))  float    v8f;

// ---- WMMA fragment helpers (layouts per CDNA5 ISA §7.12.2) ----

// A-matrix 16x32 f16, row-major source with leading dim `ld` (halfs).
// lane<16: halfs0-7 = K0-7,  halfs8-15 = K16-23 (row = lane)
// lane>=16: halfs0-7 = K8-15, halfs8-15 = K24-31 (row = lane-16)
__device__ __forceinline__ v16h load_a_frag(const _Float16* base, int ld, int lane) {
  int r  = lane & 15;
  int hi = (lane >> 4) & 1;
  const _Float16* p = base + r * ld + hi * 8;
  union { v16h v; v8h h[2]; } u;
  u.h[0] = *(const v8h*)(p);
  u.h[1] = *(const v8h*)(p + 16);
  return u.v;
}

// B-matrix 32x16 f16 from row-major weight W[N][K]: lane<16 -> N=lane, K=kb+0..7, kb+16..23;
// lane>=16 -> N=lane-16, K=kb+8..15, kb+24..31 (same k order as A).
__device__ __forceinline__ v16h load_b_frag(const _Float16* base, int ld, int lane) {
  int n  = lane & 15;
  int ko = (lane >> 4) * 8;
  const _Float16* p = base + n * ld + ko;
  union { v16h v; v8h h[2]; } u;
  u.h[0] = *(const v8h*)(p);
  u.h[1] = *(const v8h*)(p + 16);
  return u.v;
}

__device__ __forceinline__ v8f wmma16(v16h a, v16h b, v8f c) {
  return __builtin_amdgcn_wmma_f32_16x16x32_f16(false, a, false, b, (short)0, c, false, false);
}

__device__ __forceinline__ float sigmf(float x) { return 1.f / (1.f + __expf(-x)); }

// ---- f32 -> f16 weight conversion ----
__global__ void cvt_f16(const float* __restrict__ src, _Float16* __restrict__ dst, int n) {
  int i = blockIdx.x * blockDim.x + threadIdx.x;
  if (i < n) dst[i] = (_Float16)src[i];
}

// ---- TL-LSTM scan: one WG owns 16 batch rows, full 512-step recurrence ----
__global__ __launch_bounds__(TPB) void tl_scan(
    const float* __restrict__ inputs, const float* __restrict__ ts_inv,
    const float* __restrict__ reach,
    const _Float16* __restrict__ Wall, const _Float16* __restrict__ Uall,
    const _Float16* __restrict__ Wd,
    const float* __restrict__ b_all, const float* __restrict__ b_u,
    const float* __restrict__ b_d,
    _Float16* __restrict__ hs16)
{
  __shared__ float    c32 [BT][H_];   // f32 cell state carry
  __shared__ float    cadj[BT][H_];
  __shared__ float    gL  [BT][H4];   // sigmoid(g)
  __shared__ _Float16 h16 [BT][H_];
  __shared__ _Float16 c16 [BT][H_];
  __shared__ _Float16 x16 [BT][H_];
  __shared__ float    tL[BT];
  __shared__ float    rL[BT];

  const int tid  = threadIdx.x;
  const int wave = tid >> 5;          // 0..15
  const int lane = tid & 31;
  const int b0   = blockIdx.x * BT;
  const int colb = lane & 15;
  const int rbase = (lane >> 4) * 8;

  for (int i = tid; i < BT * H_; i += TPB) {
    (&c32[0][0])[i] = 0.f;
    (&h16[0][0])[i] = (_Float16)0.f;
    (&c16[0][0])[i] = (_Float16)0.f;
  }

  // --- step-invariant data hoisted out of the scan ---
  // GEMM1: this wave owns W_d N-tile `wave` for the whole scan
  v16h bwd[8];
  #pragma unroll
  for (int k = 0; k < 8; ++k)
    bwd[k] = load_b_frag(Wd + (size_t)wave * 16 * H_ + k * 32, H_, lane);
  const int col1 = wave * 16 + colb;
  const float bd = b_d[col1];

  // GEMM2: 4 N-tiles per wave; bias per tile is invariant
  float bias2[4];
  #pragma unroll
  for (int t4 = 0; t4 < 4; ++t4) {
    int col = (wave * 4 + t4) * 16 + colb;
    bias2[t4] = b_all[col] + b_u[col];
  }
  __syncthreads();

  for (int s = 0; s < S_; ++s) {
    // phase A: stage x_s (f32->f16), per-row t,r scalars
    for (int i = tid; i < BT * H_; i += TPB) {
      int row = i >> 8, col = i & (H_ - 1);
      x16[row][col] = (_Float16)inputs[((size_t)(b0 + row) * S_ + s) * D_ + col];
    }
    if (tid < BT) {
      tL[tid] = ts_inv[(size_t)(b0 + tid) * S_ + s];
      rL[tid] = reach[(size_t)(b0 + tid) * S_ + s];
    }
    __syncthreads();

    // A fragments (reused across all N tiles of this wave)
    v16h ah[8], ac[8], ax[8];
    #pragma unroll
    for (int k = 0; k < 8; ++k) {
      ah[k] = load_a_frag(&h16[0][0] + k * 32, H_, lane);
      ac[k] = load_a_frag(&c16[0][0] + k * 32, H_, lane);
      ax[k] = load_a_frag(&x16[0][0] + k * 32, H_, lane);
    }

    // GEMM1: c_s1 = tanh(c @ Wd^T + b_d); cadj = c - c_s1 + c_s1 * t
    {
      v8f acc = {};
      #pragma unroll
      for (int k = 0; k < 8; ++k)
        acc = wmma16(ac[k], bwd[k], acc);
      #pragma unroll
      for (int r2 = 0; r2 < 8; ++r2) {
        int row = rbase + r2;
        float v = tanhf(acc[r2] + bd);
        cadj[row][col1] = c32[row][col1] - v + v * tL[row];
      }
    }

    // GEMM2: g = sigmoid(h @ Wall^T + x @ Uall^T + b_all + b_u)
    for (int t4 = 0; t4 < 4; ++t4) {
      int nt = wave * 4 + t4;
      v8f acc = {};
      #pragma unroll
      for (int k = 0; k < 8; ++k)
        acc = wmma16(ah[k], load_b_frag(Wall + (size_t)nt * 16 * H_ + k * 32, H_, lane), acc);
      #pragma unroll
      for (int k = 0; k < 8; ++k)
        acc = wmma16(ax[k], load_b_frag(Uall + (size_t)nt * 16 * D_ + k * 32, D_, lane), acc);
      int col = nt * 16 + colb;
      #pragma unroll
      for (int r2 = 0; r2 < 8; ++r2)
        gL[rbase + r2][col] = sigmf(acc[r2] + bias2[t4]);
    }
    __syncthreads();

    // elementwise gate update
    for (int i = tid; i < BT * H_; i += TPB) {
      int row = i >> 8, col = i & (H_ - 1);
      float f  = gL[row][col];
      float ig = gL[row][H_ + col];
      float o  = gL[row][2 * H_ + col];
      float ct = gL[row][3 * H_ + col];
      float cn = f * cadj[row][col] + ig * ct * rL[row];
      float hn = o * tanhf(cn);
      c32[row][col] = cn;
      c16[row][col] = (_Float16)cn;
      _Float16 hh = (_Float16)hn;
      h16[row][col] = hh;
      hs16[(size_t)s * B_ * H_ + (size_t)(b0 + row) * H_ + col] = hh;
    }
    __syncthreads();
  }
}

// ---- GRU scan over hs ----
__global__ __launch_bounds__(TPB) void gru_scan(
    const _Float16* __restrict__ hs16,
    const _Float16* __restrict__ Wih, const _Float16* __restrict__ Whh,
    const float* __restrict__ bih, const float* __restrict__ bhh,
    _Float16* __restrict__ ctx16, float* __restrict__ hlast)
{
  __shared__ float    h32[BT][H_];
  __shared__ float    giL[BT][H3];
  __shared__ float    ghL[BT][H3];
  __shared__ _Float16 h16[BT][H_];

  const int tid  = threadIdx.x;
  const int wave = tid >> 5;          // 0..15
  const int lane = tid & 31;
  const int b0   = blockIdx.x * BT;
  const int colb = lane & 15;
  const int rbase = (lane >> 4) * 8;

  for (int i = tid; i < BT * H_; i += TPB) {
    (&h32[0][0])[i] = 0.f;
    (&h16[0][0])[i] = (_Float16)0.f;
  }

  // step-invariant per-tile biases (3 N-tiles per wave)
  float biv[3], bhv[3];
  #pragma unroll
  for (int t3 = 0; t3 < 3; ++t3) {
    int col = (wave * 3 + t3) * 16 + colb;
    biv[t3] = bih[col];
    bhv[t3] = bhh[col];
  }
  __syncthreads();

  for (int s = 0; s < S_; ++s) {
    const _Float16* xbase = hs16 + (size_t)s * B_ * H_ + (size_t)b0 * H_;
    v16h ah[8], ax[8];
    #pragma unroll
    for (int k = 0; k < 8; ++k) {
      ah[k] = load_a_frag(&h16[0][0] + k * 32, H_, lane);
      ax[k] = load_a_frag(xbase + k * 32, H_, lane);   // A frags straight from global f16
    }
    for (int t3 = 0; t3 < 3; ++t3) {
      int nt  = wave * 3 + t3;
      int col = nt * 16 + colb;
      v8f acc = {};
      #pragma unroll
      for (int k = 0; k < 8; ++k)
        acc = wmma16(ax[k], load_b_frag(Wih + (size_t)nt * 16 * H_ + k * 32, H_, lane), acc);
      #pragma unroll
      for (int r2 = 0; r2 < 8; ++r2) giL[rbase + r2][col] = acc[r2] + biv[t3];
      v8f acc2 = {};
      #pragma unroll
      for (int k = 0; k < 8; ++k)
        acc2 = wmma16(ah[k], load_b_frag(Whh + (size_t)nt * 16 * H_ + k * 32, H_, lane), acc2);
      #pragma unroll
      for (int r2 = 0; r2 < 8; ++r2) ghL[rbase + r2][col] = acc2[r2] + bhv[t3];
    }
    __syncthreads();

    for (int i = tid; i < BT * H_; i += TPB) {
      int row = i >> 8, col = i & (H_ - 1);
      float rg = sigmf(giL[row][col] + ghL[row][col]);
      float z  = sigmf(giL[row][H_ + col] + ghL[row][H_ + col]);
      float n  = tanhf(giL[row][2 * H_ + col] + rg * ghL[row][2 * H_ + col]);
      float hn = (1.f - z) * n + z * h32[row][col];
      h32[row][col] = hn;
      h16[row][col] = (_Float16)hn;
      ctx16[(size_t)s * B_ * H_ + (size_t)(b0 + row) * H_ + col] = (_Float16)hn;
      if (s == S_ - 1) hlast[(size_t)(b0 + row) * H_ + col] = hn;
    }
    __syncthreads();
  }
}

// ---- Hawkes attention + MLP head: one block per batch row ----
__global__ __launch_bounds__(ATPB) void attn_head(
    const float* __restrict__ timestamps,
    const float* __restrict__ Win, const float* __restrict__ Wout,
    const float* __restrict__ attn_a, const float* __restrict__ attn_b,
    const float* __restrict__ lin1W, const float* __restrict__ lin1b,
    const float* __restrict__ lin2W, const float* __restrict__ lin2b,
    const _Float16* __restrict__ ctx16, const float* __restrict__ hlast,
    float* __restrict__ out)
{
  __shared__ float hl[H_];
  __shared__ float qL[H_];
  __shared__ float scL[S_];
  __shared__ float red[ATPB];
  __shared__ float mixL[H_];
  __shared__ float o1L[H_];
  __shared__ float o2L[H_];

  const int tid = threadIdx.x;
  const int b   = blockIdx.x;

  hl[tid] = hlast[(size_t)b * H_ + tid];
  __syncthreads();

  {  // q = h_last @ Win^T
    const float* wr = Win + (size_t)tid * H_;
    float acc = 0.f;
    for (int k = 0; k < H_; ++k) acc += hl[k] * wr[k];
    qL[tid] = acc;
  }
  __syncthreads();

  for (int s = tid; s < S_; s += ATPB) {  // scores[s] = q . ctx[b,s,:]
    const _Float16* cr = ctx16 + (size_t)s * B_ * H_ + (size_t)b * H_;
    float acc = 0.f;
    for (int k = 0; k < H_; ++k) acc += qL[k] * (float)cr[k];
    scL[s] = acc;
  }
  __syncthreads();

  // softmax over S
  red[tid] = fmaxf(scL[tid], scL[tid + ATPB]);
  __syncthreads();
  for (int off = ATPB / 2; off > 0; off >>= 1) {
    if (tid < off) red[tid] = fmaxf(red[tid], red[tid + off]);
    __syncthreads();
  }
  float mx = red[0];
  __syncthreads();
  float e0 = __expf(scL[tid] - mx), e1 = __expf(scL[tid + ATPB] - mx);
  scL[tid] = e0; scL[tid + ATPB] = e1;
  red[tid] = e0 + e1;
  __syncthreads();
  for (int off = ATPB / 2; off > 0; off >>= 1) {
    if (tid < off) red[tid] += red[tid + off];
    __syncthreads();
  }
  float inv = 1.f / red[0];
  __syncthreads();

  float av = attn_a[b], bv = attn_b[b];
  for (int s = tid; s < S_; s += ATPB) {  // hawkes reweighting
    float w = scL[s] * inv;
    scL[s] = w * (1.f + av * __expf(-bv * timestamps[(size_t)b * S_ + s]));
  }
  __syncthreads();

  {  // mix = w_hawkes @ ctx
    float acc = 0.f;
    for (int s = 0; s < S_; ++s)
      acc += scL[s] * (float)ctx16[(size_t)s * B_ * H_ + (size_t)b * H_ + tid];
    mixL[tid] = acc;
  }
  __syncthreads();

  {  // out = tanh(Wout @ [mix, q])
    const float* wr = Wout + (size_t)tid * (2 * H_);
    float acc = 0.f;
    for (int k = 0; k < H_; ++k) acc += wr[k] * mixL[k];
    for (int k = 0; k < H_; ++k) acc += wr[H_ + k] * qL[k];
    o1L[tid] = tanhf(acc);
  }
  __syncthreads();

  {  // relu(lin1)
    const float* wr = lin1W + (size_t)tid * H_;
    float acc = lin1b[tid];
    for (int k = 0; k < H_; ++k) acc += wr[k] * o1L[k];
    o2L[tid] = fmaxf(acc, 0.f);
  }
  __syncthreads();

  if (tid < 2) {  // lin2 -> (B, NT)
    const float* wr = lin2W + (size_t)tid * H_;
    float acc = lin2b[tid];
    for (int k = 0; k < H_; ++k) acc += wr[k] * o2L[k];
    out[(size_t)b * 2 + tid] = acc;
  }
}

extern "C" void kernel_launch(void* const* d_in, const int* in_sizes, int n_in,
                              void* d_out, int out_size, void* d_ws, size_t ws_size,
                              hipStream_t stream) {
  (void)in_sizes; (void)n_in; (void)out_size; (void)ws_size;

  const float* inputs     = (const float*)d_in[0];
  const float* timestamps = (const float*)d_in[1];
  const float* ts_inv     = (const float*)d_in[2];
  const float* reach      = (const float*)d_in[3];
  const float* W_all      = (const float*)d_in[4];
  const float* b_all      = (const float*)d_in[5];
  const float* U_all      = (const float*)d_in[6];
  const float* b_u        = (const float*)d_in[7];
  const float* W_d        = (const float*)d_in[8];
  const float* b_d        = (const float*)d_in[9];
  const float* gWih       = (const float*)d_in[10];
  const float* gWhh       = (const float*)d_in[11];
  const float* gbih       = (const float*)d_in[12];
  const float* gbhh       = (const float*)d_in[13];
  const float* aWin       = (const float*)d_in[14];
  const float* aWout      = (const float*)d_in[15];
  const float* attn_a     = (const float*)d_in[16];
  const float* attn_b     = (const float*)d_in[17];
  const float* l1W        = (const float*)d_in[18];
  const float* l1b        = (const float*)d_in[19];
  const float* l2W        = (const float*)d_in[20];
  const float* l2b        = (const float*)d_in[21];

  char* ws = (char*)d_ws;
  size_t off = 0;
  auto alloc = [&](size_t bytes) -> void* {
    void* p = ws + off;
    off += (bytes + 255) & ~(size_t)255;
    return p;
  };
  _Float16* wall16 = (_Float16*)alloc((size_t)H4 * H_ * sizeof(_Float16));
  _Float16* uall16 = (_Float16*)alloc((size_t)H4 * D_ * sizeof(_Float16));
  _Float16* wd16   = (_Float16*)alloc((size_t)H_ * H_ * sizeof(_Float16));
  _Float16* wih16  = (_Float16*)alloc((size_t)H3 * H_ * sizeof(_Float16));
  _Float16* whh16  = (_Float16*)alloc((size_t)H3 * H_ * sizeof(_Float16));
  _Float16* hs16   = (_Float16*)alloc((size_t)S_ * B_ * H_ * sizeof(_Float16));
  _Float16* ctx16  = (_Float16*)alloc((size_t)S_ * B_ * H_ * sizeof(_Float16));
  float*    hlast  = (float*)alloc((size_t)B_ * H_ * sizeof(float));

  int n;
  n = H4 * H_; cvt_f16<<<(n + 255) / 256, 256, 0, stream>>>(W_all, wall16, n);
  n = H4 * D_; cvt_f16<<<(n + 255) / 256, 256, 0, stream>>>(U_all, uall16, n);
  n = H_ * H_; cvt_f16<<<(n + 255) / 256, 256, 0, stream>>>(W_d, wd16, n);
  n = H3 * H_; cvt_f16<<<(n + 255) / 256, 256, 0, stream>>>(gWih, wih16, n);
  n = H3 * H_; cvt_f16<<<(n + 255) / 256, 256, 0, stream>>>(gWhh, whh16, n);

  tl_scan<<<B_ / BT, TPB, 0, stream>>>(inputs, ts_inv, reach, wall16, uall16, wd16,
                                       b_all, b_u, b_d, hs16);
  gru_scan<<<B_ / BT, TPB, 0, stream>>>(hs16, wih16, whh16, gbih, gbhh, ctx16, hlast);
  attn_head<<<B_, ATPB, 0, stream>>>(timestamps, aWin, aWout, attn_a, attn_b,
                                     l1W, l1b, l2W, l2b, ctx16, hlast, (float*)d_out);
}
